// Center_loss_9388798509687
// MI455X (gfx1250) — matrix operand order repaired and checked by reading, not compiled, not run on hardware
//
#include <hip/hip_runtime.h>
#include <hip/hip_bf16.h>

// ---------------------------------------------------------------------------
// Center loss, fused single pass:
//   loss = ( sum_i ||x_i||^2  -  sum_c ||S_c||^2 / max(count_c,1) ) / (n*d)
// Kernel 1: zero workspace.
// Kernel 2: stream features once; per-block LDS class sums (ds_add_f32,
//           bank-conflict-free), global f32 atomic flush, f64 atomic for sumsq.
// Kernel 3: Gram-diagonal of S (64x256) via v_wmma_f32_16x16x32_f16, finalize.
// ---------------------------------------------------------------------------

#define D_DIM   256
#define NCLASS  64
#define NBLK    512
#define NTHR    256           // 8 wave32
#define SUMS_F  (NCLASS * D_DIM)      // 16384 floats
// ws layout: [0..15] double accum[2]; then float gsums[16384]; float gcnt[64]
#define WS_FLOATS (4 + SUMS_F + NCLASS)   // doubles counted as 4 floats

typedef _Float16 v16h __attribute__((ext_vector_type(16)));
typedef float    v8f  __attribute__((ext_vector_type(8)));

__global__ void center_zero(float* ws) {
    int i = blockIdx.x * NTHR + threadIdx.x;
    if (i < WS_FLOATS) ws[i] = 0.0f;
}

__global__ void __launch_bounds__(NTHR)
center_main(const float* __restrict__ feat, const int* __restrict__ labels,
            int n, double* __restrict__ accum,
            float* __restrict__ gsums, float* __restrict__ gcnt) {
    extern __shared__ float smem[];
    float* lsum = smem;            // 64*256 per-block class sums
    float* lcnt = smem + SUMS_F;   // 64 class counts (reused as reduce scratch)

    const int tid  = threadIdx.x;
    const int lane = tid & 31;
    const int wave = tid >> 5;

    for (int i = tid; i < SUMS_F; i += NTHR) lsum[i] = 0.0f;
    if (tid < NCLASS) lcnt[tid] = 0.0f;
    __syncthreads();

    const int gw     = blockIdx.x * (NTHR / 32) + wave;
    const int nwaves = gridDim.x * (NTHR / 32);

    float sumsq = 0.0f;
    for (int row = gw; row < n; row += nwaves) {
        const int lbl = labels[row];                 // uniform per wave -> bcast
        const float* r = feat + (size_t)row * D_DIM;
        float* dst = lsum + lbl * D_DIM;
#pragma unroll
        for (int c = 0; c < 8; ++c) {
            // dim = lane + 32c: coalesced 128B load, 32 distinct LDS banks
            float v = __builtin_nontemporal_load(r + lane + 32 * c);
            sumsq = fmaf(v, v, sumsq);
            atomicAdd(dst + lane + 32 * c, v);       // ds_add_f32
        }
        if (lane == 0) atomicAdd(lcnt + lbl, 1.0f);
    }
    __syncthreads();

    // flush block-local sums/counts to global with f32 atomics
    for (int i = tid; i < SUMS_F; i += NTHR) {
        float v = lsum[i];
        if (v != 0.0f) atomicAdd(gsums + i, v);
    }
    if (tid < NCLASS) {
        float v = lcnt[tid];
        if (v != 0.0f) atomicAdd(gcnt + tid, v);
    }

    // wave-level reduce of sumsq, then one f64 atomic per block
#pragma unroll
    for (int off = 16; off > 0; off >>= 1)
        sumsq += __shfl_down(sumsq, off, 32);
    __syncthreads();                    // lcnt flush reads done; safe to reuse
    if (lane == 0) lcnt[wave] = sumsq;
    __syncthreads();
    if (tid == 0) {
        float t = 0.0f;
        for (int w = 0; w < NTHR / 32; ++w) t += lcnt[w];
        atomicAdd(accum, (double)t);    // global_atomic_add_f64
    }
}

// Finalize: term2 = sum_c ||S_c||^2 / max(cnt_c,1) via Gram diagonal in WMMA.
// 4 waves, wave wv owns classes [16*wv, 16*wv+16); 8 k-steps of K=32.
__global__ void __launch_bounds__(128)
center_final(const double* __restrict__ accum, const float* __restrict__ gsums,
             const float* __restrict__ gcnt, float* __restrict__ out,
             int n, int d) {
    __shared__ float red[128];
    const int tid  = threadIdx.x;
    const int lane = tid & 31;
    const int wv   = tid >> 5;          // 0..3
    const int c0   = wv * 16;
    const int m    = lane & 15;         // A row / B col for this lane
    const int h    = lane >> 4;         // lane half

    v8f acc = {};
#pragma unroll
    for (int kk = 0; kk < D_DIM; kk += 32) {
        v16h a, b;
#pragma unroll
        for (int i = 0; i < 16; ++i) {
            // A (16x32 f16): VGPR v holds K = 8h + i (i<8) | 16 + 8h + (i-8)
            int ka = kk + ((i < 8) ? (8 * h + i) : (16 + 8 * h + (i - 8)));
            a[i] = (_Float16)gsums[(c0 + m) * D_DIM + ka];
            // B (32x16 f16): VGPR v holds K = 8h + i (i<8) | 16 + 8h + (i-8); col = m
            int kb = kk + ((i < 8) ? (8 * h + i) : (16 + 8 * h + (i - 8)));
            b[i] = (_Float16)gsums[(c0 + m) * D_DIM + kb];
        }
        acc = __builtin_amdgcn_wmma_f32_16x16x32_f16(
            /*neg_a=*/false, a, /*neg_b=*/false, b,
            /*c_mod=*/(short)0, acc, /*reuse_a=*/false, /*reuse_b=*/false);
    }

    // Diagonal of D: (M,N) -> VGPR r with M = r + 8h, N = lane&15.
    // Exactly one r matches per diagonal-carrying lane; select branch-free,
    // then a single division.
    float diagv = 0.0f;
    bool  hasd  = false;
#pragma unroll
    for (int r = 0; r < 8; ++r) {
        bool sel = (r + 8 * h) == m;     // v_cndmask chain, no exec branches
        diagv = sel ? acc[r] : diagv;
        hasd  = hasd | sel;
    }
    float cnt = gcnt[c0 + m];
    cnt = (cnt < 1.0f) ? 1.0f : cnt;
    float part = hasd ? (diagv / cnt) : 0.0f;

    red[tid] = part;
    __syncthreads();
    if (tid == 0) {
        double t2 = 0.0;
        for (int i = 0; i < 128; ++i) t2 += (double)red[i];
        double loss = (accum[0] - t2) / ((double)n * (double)d);
        out[0] = (float)loss;
    }
}

extern "C" void kernel_launch(void* const* d_in, const int* in_sizes, int n_in,
                              void* d_out, int out_size, void* d_ws, size_t ws_size,
                              hipStream_t stream) {
    const float* feat   = (const float*)d_in[0];
    const int*   labels = (const int*)d_in[1];
    const int n = in_sizes[1];                 // 262144
    const int d = in_sizes[0] / n;             // 256

    double* accum = (double*)d_ws;
    float*  gsums = (float*)((char*)d_ws + 16);
    float*  gcnt  = gsums + SUMS_F;

    center_zero<<<(WS_FLOATS + NTHR - 1) / NTHR, NTHR, 0, stream>>>((float*)d_ws);
    center_main<<<NBLK, NTHR, (SUMS_F + NCLASS) * sizeof(float), stream>>>(
        feat, labels, n, accum, gsums, gcnt);
    center_final<<<1, 128, 0, stream>>>(accum, gsums, gcnt, (float*)d_out, n, d);
}
